// BTSPMemoryBank_13486197309497
// MI455X (gfx1250) — matrix operand order repaired and checked by reading, not compiled, not run on hardware
//
#include <hip/hip_runtime.h>
#include <hip/hip_bf16.h>
#include <cstdint>

// B=4096, D=1024, N=65536, K=32
#define B_ROWS 4096
#define DIM    1024
#define N_MEM  65536
#define TOPK   32
#define EPS_N  1e-12f

typedef __attribute__((ext_vector_type(16))) __bf16 v16bf;
typedef __attribute__((ext_vector_type(8)))  __bf16 v8bf;
typedef __attribute__((ext_vector_type(8)))  float  v8f;
typedef __attribute__((ext_vector_type(2)))  float  v2f;

union Frag16 { v16bf v; v8bf h[2]; };
union Pack4  { __bf16 h[4]; unsigned long long u; };

// ---------------------------------------------------------------------------
// Kernel 0: convert fp32 memory bank -> bf16 (128 MB, fits in 192 MB L2)
// ---------------------------------------------------------------------------
__global__ void k_cvt_mem(const float4* __restrict__ src,
                          unsigned long long* __restrict__ dst) {
  int i = blockIdx.x * blockDim.x + threadIdx.x;
  float4 f = src[i];
  Pack4 p;
  p.h[0] = (__bf16)f.x; p.h[1] = (__bf16)f.y;
  p.h[2] = (__bf16)f.z; p.h[3] = (__bf16)f.w;
  dst[i] = p.u;
}

// ---------------------------------------------------------------------------
// Kernel 1: proj = query @ W_key^T in fp32 via V_WMMA_F32_16X16X4_F32.
// proj[b][d] = sum_k Q[b][k] * W[d][k]  (both row-major, k contiguous).
// A 16x4 f32 frag: m=lane%16, k = 2*(lane/16)+{0,1}.  B 4x16: n=lane%16, same k.
// 4 accumulators to break the WMMA dependence chain.
// ---------------------------------------------------------------------------
__global__ void k_proj(const float* __restrict__ Q, const float* __restrict__ W,
                       float* __restrict__ P) {
  const int tid = threadIdx.x, lane = tid & 31, wave = tid >> 5;
  const int tileId  = blockIdx.x * 8 + wave;     // 16384 tiles total
  const int rowTile = tileId >> 6;               // DIM/16 = 64 col-tiles
  const int colTile = tileId & 63;
  const int mr   = lane & 15;
  const int koff = (lane >> 4) * 2;
  const float* qrow = Q + (size_t)(rowTile * 16 + mr) * DIM + koff;
  const float* wrow = W + (size_t)(colTile * 16 + mr) * DIM + koff;

  v8f c0 = {}, c1 = {}, c2 = {}, c3 = {};
  for (int kb = 0; kb < DIM; kb += 16) {
    v2f a0 = *(const v2f*)(qrow + kb +  0), b0 = *(const v2f*)(wrow + kb +  0);
    v2f a1 = *(const v2f*)(qrow + kb +  4), b1 = *(const v2f*)(wrow + kb +  4);
    v2f a2 = *(const v2f*)(qrow + kb +  8), b2 = *(const v2f*)(wrow + kb +  8);
    v2f a3 = *(const v2f*)(qrow + kb + 12), b3 = *(const v2f*)(wrow + kb + 12);
    c0 = __builtin_amdgcn_wmma_f32_16x16x4_f32(false, a0, false, b0, (short)0, c0, false, false);
    c1 = __builtin_amdgcn_wmma_f32_16x16x4_f32(false, a1, false, b1, (short)0, c1, false, false);
    c2 = __builtin_amdgcn_wmma_f32_16x16x4_f32(false, a2, false, b2, (short)0, c2, false, false);
    c3 = __builtin_amdgcn_wmma_f32_16x16x4_f32(false, a3, false, b3, (short)0, c3, false, false);
  }
  v8f c = (c0 + c1) + (c2 + c3);
  const int rBase = rowTile * 16 + ((lane >> 4) << 3);
  const int col   = colTile * 16 + (lane & 15);
#pragma unroll
  for (int r = 0; r < 8; ++r)
    P[(size_t)(rBase + r) * DIM + col] = c[r];
}

// ---------------------------------------------------------------------------
// Kernel 2: per-row L2 norm of proj, scale, convert to bf16 (qn).
// One wave (32 lanes) per row; normalization folded in BEFORE bf16 conversion.
// ---------------------------------------------------------------------------
__global__ void k_norm(const float* __restrict__ P,
                       unsigned long long* __restrict__ QN) {
  const int tid = threadIdx.x, lane = tid & 31, wave = tid >> 5;
  const int row = blockIdx.x * 8 + wave;
  const float4* p4 = (const float4*)(P + (size_t)row * DIM);
  float4 v[8];
  float s = 0.f;
#pragma unroll
  for (int j = 0; j < 8; ++j) {
    v[j] = p4[lane + 32 * j];
    s += v[j].x * v[j].x + v[j].y * v[j].y + v[j].z * v[j].z + v[j].w * v[j].w;
  }
#pragma unroll
  for (int off = 16; off >= 1; off >>= 1) s += __shfl_xor(s, off, 32);
  const float inv = 1.0f / fmaxf(sqrtf(s), EPS_N);
  unsigned long long* o = QN + (size_t)row * (DIM / 4);
#pragma unroll
  for (int j = 0; j < 8; ++j) {
    Pack4 p;
    p.h[0] = (__bf16)(v[j].x * inv); p.h[1] = (__bf16)(v[j].y * inv);
    p.h[2] = (__bf16)(v[j].z * inv); p.h[3] = (__bf16)(v[j].w * inv);
    o[lane + 32 * j] = p.u;
  }
}

// ---------------------------------------------------------------------------
// Kernel 3: fused sim-GEMM (bf16 WMMA, 2x2 tiles/wave) + streaming top-32
//           + softmax + fp32 gather.
// Per k-step: 4 WMMA : 4 ds_load_b128 : 4 global_load_b128  (1:1:1)
// ---------------------------------------------------------------------------
#define CAP 320                        // candidate slots per row
// LDS: As 64K | tV 4K | tI 4K | thr 128 | cV 40960 | cI 40960 | cN 128
#define OFF_TV  65536
#define OFF_TI  69632
#define OFF_THR 73728
#define OFF_CV  73856
#define OFF_CI  114816
#define OFF_CN  155776
#define SM_TOTAL 155904

__device__ __forceinline__ void merge_row(float* tv, int* ti, const float* cv,
                                          const int* ci, int cnt, float* thr) {
  float mv = tv[0]; int mp = 0;
  for (int j = 1; j < 32; ++j) { float t = tv[j]; if (t < mv) { mv = t; mp = j; } }
  for (int c = 0; c < cnt; ++c) {
    float v = cv[c];
    if (v > mv) {
      tv[mp] = v; ti[mp] = ci[c];
      mv = tv[0]; mp = 0;
      for (int j = 1; j < 32; ++j) { float t = tv[j]; if (t < mv) { mv = t; mp = j; } }
    }
  }
  *thr = mv;
}

__global__ void __launch_bounds__(256, 1)
k_sim_topk(const __bf16* __restrict__ QN, const __bf16* __restrict__ MB,
           const float* __restrict__ MF, float* __restrict__ outRet,
           float* __restrict__ outSim) {
  extern __shared__ char smem[];
  __bf16* As  = (__bf16*)smem;                   // 32 rows x 1024 bf16
  float*  tV  = (float*)(smem + OFF_TV);
  int*    tI  = (int*)  (smem + OFF_TI);
  float*  thr = (float*)(smem + OFF_THR);
  float*  cV  = (float*)(smem + OFF_CV);
  int*    cI  = (int*)  (smem + OFF_CI);
  int*    cN  = (int*)  (smem + OFF_CN);

  const int tid = threadIdx.x, lane = tid & 31, wave = tid >> 5;
  const int rowBase = blockIdx.x * 32;

  { // stage 32 query rows (contiguous 64 KB) into LDS
    const uint4* src = (const uint4*)(QN + (size_t)rowBase * DIM);
    uint4* dst = (uint4*)As;
#pragma unroll
    for (int i = 0; i < 16; ++i) dst[tid + 256 * i] = src[tid + 256 * i];
  }
  if (tid < 32) {
    thr[tid] = -INFINITY; cN[tid] = 0;
    for (int j = 0; j < 32; ++j) { tV[tid * 32 + j] = -INFINITY; tI[tid * 32 + j] = 0; }
  }
  __syncthreads();

  const int mr = lane & 15;
  const int hi = lane >> 4;
  // A frag (16-bit 16x32): element j -> k = (j/8)*16 + hi*8 + (j%8)
  const __bf16* a0base = As + (size_t)mr * DIM + hi * 8;
  const __bf16* a1base = a0base + 16 * DIM;

  for (int step = 0; step < N_MEM / 256; ++step) {
    const int col0 = step * 256 + wave * 32 + mr;   // col tile 0 of this wave
    const int col1 = col0 + 16;                     // col tile 1
    // B frag (32x16): n = lane%16, element j -> k = (j/8)*16 + hi*8 + (j%8)
    const __bf16* bbase0 = MB + (size_t)col0 * DIM + hi * 8;
    const __bf16* bbase1 = MB + (size_t)col1 * DIM + hi * 8;
    v8f c00 = {}, c01 = {}, c10 = {}, c11 = {};
    for (int kb = 0; kb < DIM; kb += 32) {
      Frag16 b0, b1, a0, a1;
      b0.h[0] = *(const v8bf*)(bbase0 + kb);  b0.h[1] = *(const v8bf*)(bbase0 + kb + 16);
      b1.h[0] = *(const v8bf*)(bbase1 + kb);  b1.h[1] = *(const v8bf*)(bbase1 + kb + 16);
      a0.h[0] = *(const v8bf*)(a0base + kb);  a0.h[1] = *(const v8bf*)(a0base + kb + 16);
      a1.h[0] = *(const v8bf*)(a1base + kb);  a1.h[1] = *(const v8bf*)(a1base + kb + 16);
      c00 = __builtin_amdgcn_wmma_f32_16x16x32_bf16(false, a0.v, false, b0.v, (short)0, c00, false, false);
      c01 = __builtin_amdgcn_wmma_f32_16x16x32_bf16(false, a0.v, false, b1.v, (short)0, c01, false, false);
      c10 = __builtin_amdgcn_wmma_f32_16x16x32_bf16(false, a1.v, false, b0.v, (short)0, c10, false, false);
      c11 = __builtin_amdgcn_wmma_f32_16x16x32_bf16(false, a1.v, false, b1.v, (short)0, c11, false, false);
    }
    const int r0 = hi * 8;
#pragma unroll
    for (int r = 0; r < 8; ++r) {
      const int lrA = r0 + r;        // rows 0..15  (row tile 0)
      const int lrB = lrA + 16;      // rows 16..31 (row tile 1)
      float v;
      v = c00[r];
      if (v > thr[lrA]) { int p = atomicAdd(&cN[lrA], 1); cV[lrA * CAP + p] = v; cI[lrA * CAP + p] = col0; }
      v = c01[r];
      if (v > thr[lrA]) { int p = atomicAdd(&cN[lrA], 1); cV[lrA * CAP + p] = v; cI[lrA * CAP + p] = col1; }
      v = c10[r];
      if (v > thr[lrB]) { int p = atomicAdd(&cN[lrB], 1); cV[lrB * CAP + p] = v; cI[lrB * CAP + p] = col0; }
      v = c11[r];
      if (v > thr[lrB]) { int p = atomicAdd(&cN[lrB], 1); cV[lrB * CAP + p] = v; cI[lrB * CAP + p] = col1; }
    }
    __syncthreads();
    if (tid < 32) {
      int cnt = cN[tid];   // flush when next step (<=256 adds) could overflow CAP=320
      if (cnt > 64) { merge_row(tV + tid * 32, tI + tid * 32, cV + tid * CAP, cI + tid * CAP, cnt, &thr[tid]); cN[tid] = 0; }
    }
    __syncthreads();
  }

  // --- epilogue: final merge, sort desc, emit top_sim, softmax weights ---
  if (tid < 32) {
    float* tv = tV + tid * 32; int* ti = tI + tid * 32;
    merge_row(tv, ti, cV + tid * CAP, cI + tid * CAP, cN[tid], &thr[tid]);
    for (int i = 0; i < 31; ++i) {            // selection sort, descending
      int best = i; float bv = tv[i];
      for (int j = i + 1; j < 32; ++j) if (tv[j] > bv) { bv = tv[j]; best = j; }
      if (best != i) {
        float fv = tv[i]; tv[i] = tv[best]; tv[best] = fv;
        int iv = ti[i]; ti[i] = ti[best]; ti[best] = iv;
      }
    }
    const int grow = rowBase + tid;
    for (int j = 0; j < 32; ++j) outSim[(size_t)grow * TOPK + j] = tv[j];
    float m = tv[0], sum = 0.f;
    for (int j = 0; j < 32; ++j) { float e = __expf(tv[j] - m); tv[j] = e; sum += e; }
    float isum = 1.0f / sum;
    for (int j = 0; j < 32; ++j) tv[j] *= isum;
  }
  __syncthreads();

  // --- gather: retrieved[row] = sum_k w_k * memory[idx_k]  (fp32 source) ---
#pragma unroll 1
  for (int rr = 0; rr < 4; ++rr) {
    const int lr = wave * 4 + rr;
    const float* tv = tV + lr * 32; const int* ti = tI + lr * 32;
    float4 acc[8];
#pragma unroll
    for (int j = 0; j < 8; ++j) acc[j] = make_float4(0.f, 0.f, 0.f, 0.f);
    for (int k = 0; k < 32; ++k) {
      const float w = tv[k];
      const float4* src = (const float4*)(MF + (size_t)ti[k] * DIM);
#pragma unroll
      for (int j = 0; j < 8; ++j) {
        float4 s = src[lane + 32 * j];
        acc[j].x += w * s.x; acc[j].y += w * s.y;
        acc[j].z += w * s.z; acc[j].w += w * s.w;
      }
    }
    float4* o = (float4*)(outRet + (size_t)(rowBase + lr) * DIM);
#pragma unroll
    for (int j = 0; j < 8; ++j) o[lane + 32 * j] = acc[j];
  }
}

// ---------------------------------------------------------------------------
// host glue
// ---------------------------------------------------------------------------
extern "C" void kernel_launch(void* const* d_in, const int* in_sizes, int n_in,
                              void* d_out, int out_size, void* d_ws, size_t ws_size,
                              hipStream_t stream) {
  (void)in_sizes; (void)n_in; (void)out_size; (void)ws_size;
  const float* query  = (const float*)d_in[0];
  const float* memory = (const float*)d_in[1];
  const float* W_key  = (const float*)d_in[2];

  float* outRet = (float*)d_out;                               // (B, D)
  float* outSim = outRet + (size_t)B_ROWS * DIM;               // (B, K)

  // workspace: memB (128 MB bf16) | qnB (8 MB bf16) | proj (16 MB f32) = 152 MB
  char* ws = (char*)d_ws;
  __bf16* memB = (__bf16*)ws;
  __bf16* qnB  = (__bf16*)(ws + (size_t)N_MEM * DIM * 2);
  float*  proj = (float*) (ws + (size_t)N_MEM * DIM * 2 + (size_t)B_ROWS * DIM * 2);

  // 0) memory fp32 -> bf16 (64M elems, 4 per thread)
  k_cvt_mem<<<dim3((N_MEM * (size_t)DIM / 4) / 256), dim3(256), 0, stream>>>(
      (const float4*)memory, (unsigned long long*)memB);

  // 1) projection: 16384 16x16 tiles, 8 per block
  k_proj<<<dim3(2048), dim3(256), 0, stream>>>(query, W_key, proj);

  // 2) normalize + bf16: one wave per row
  k_norm<<<dim3(B_ROWS / 8), dim3(256), 0, stream>>>(proj, (unsigned long long*)qnB);

  // 3) fused sim GEMM + top-k + softmax + gather: 32 rows per block
  k_sim_topk<<<dim3(B_ROWS / 32), dim3(256), SM_TOTAL, stream>>>(
      qnB, memB, memory, outRet, outSim);
}
